// HMGRUV2_89309549953830
// MI455X (gfx1250) — compile-verified, matrix operand order checked by reading, not run on hardware
//
#include <hip/hip_runtime.h>

// ---------------------------------------------------------------------------
// HMGRU (2-layer masked GRU) for gfx1250 using v_wmma_f32_16x16x32_bf16 and
// global_load_async_to_lds_b128 DMA staging of the x tiles.
// B=256 T=256 IN=256 H=256, 3H=768 gates.
// Grid: 16 blocks (batch tiles of 16 rows). Block: 512 threads = 16 waves.
// Wave w owns hidden columns [16w, 16w+16): r/z fused K=768 GEMM, n-gate kept
// as separate gi_n (K=512) and gh_n (K=256) GEMMs (n = tanh(gi + r*gh)).
// Weights prepacked to bf16 in workspace in WMMA-B-fragment order (L2-resident).
// ---------------------------------------------------------------------------

#define BDIM   256
#define TDIM   256
#define INDIM  256
#define HDIM   256
#define KTOT   768            // [x(256) | h_other(256) | h_self(256)]
#define XSTRIDE 776           // 768 + 8 pad (1552B rows: bank spread, 16B mult)
#define HSTRIDE 264           // 256 + 8 pad

typedef __bf16 bf16_t;
typedef __attribute__((ext_vector_type(16))) __bf16 v16bf;
typedef __attribute__((ext_vector_type(8)))  __bf16 v8bf;
typedef __attribute__((ext_vector_type(4)))  __bf16 v4bf;
typedef __attribute__((ext_vector_type(8)))  float  v8f;
typedef __attribute__((ext_vector_type(4)))  float  v4f;

static constexpr int W_RZ  = 512 * 768;   // fused [W_ih;W_hh] rows 0..511 (r,z)
static constexpr int W_GIN = 256 * 512;   // W_ih rows 512..767 (n gate, input path)
static constexpr int W_GHN = 256 * 256;   // W_hh rows 512..767 (n gate, hidden path)
static constexpr int WL    = W_RZ + W_GIN + W_GHN;  // per-layer packed elems

// Load one 16x32 bf16 WMMA fragment for this lane: two contiguous 16B chunks
// 32B apart (ISA 16-bit A/B layout). Caller bakes lane row/col + khalf into p.
__device__ __forceinline__ v16bf ld_frag(const bf16_t* p) {
  v8bf lo = *(const v8bf*)(p);
  v8bf hi = *(const v8bf*)(p + 16);
  v16bf r;
#pragma unroll
  for (int i = 0; i < 8; ++i) { r[i] = lo[i]; r[i + 8] = hi[i]; }
  return r;
}

__device__ __forceinline__ v8f wmma_bf16(v16bf a, v16bf b, v8f c) {
  return __builtin_amdgcn_wmma_f32_16x16x32_bf16(false, a, false, b,
                                                 (short)0, c, false, false);
}

__device__ __forceinline__ float sigmoidf_(float x) {
  return 1.0f / (1.0f + __expf(-x));
}

// Async DMA one 16x256 f32 tile (16KB) global -> LDS. 1024 16B chunks,
// 2 per thread. Tracked on ASYNCcnt; pair with wait_async()+barrier.
__device__ __forceinline__ void issue_x_tile(const float* __restrict__ xsrc,
                                             int b0row, int t,
                                             const float* xraw_s, int tid) {
  const char* gbase = (const char*)xsrc;
#pragma unroll
  for (int j = 0; j < 2; ++j) {
    int c = tid + j * 512;
    int m = c >> 6;                 // 64 chunks per row
    int o = (c & 63) << 4;          // byte offset within row
    unsigned long long g = (unsigned long long)
        (gbase + ((size_t)(b0row + m) * TDIM + t) * (INDIM * 4) + o);
    unsigned int l = (unsigned int)(size_t)((const char*)xraw_s + m * 1024 + o);
    asm volatile("global_load_async_to_lds_b128 %0, %1, off"
                 :: "v"(l), "v"(g) : "memory");
  }
}

__device__ __forceinline__ void wait_async() {
  asm volatile("s_wait_asynccnt 0" ::: "memory");
}

// Stage xin[16][KTOT] = [xraw*sx | h_other*so | h_self*ss], 4-wide.
__device__ __forceinline__ void stage_xin(
    bf16_t* __restrict__ xin, const float* __restrict__ xraw,
    const bf16_t* __restrict__ h_oth, const bf16_t* __restrict__ h_self,
    const float* __restrict__ sx, const float* __restrict__ so,
    const float* __restrict__ ss, int tid) {
#pragma unroll 2
  for (int e4 = tid; e4 < 3072; e4 += 512) {   // 16 rows * 192 vec4
    int m = e4 / 192;
    int k = (e4 - m * 192) * 4;
    v4bf o;
    if (k < 256) {
      v4f xv = *(const v4f*)(xraw + m * 256 + k);
      float s = sx[m];
#pragma unroll
      for (int i = 0; i < 4; ++i) o[i] = (bf16_t)(xv[i] * s);
    } else if (k < 512) {
      v4bf hv = *(const v4bf*)(h_oth + (size_t)m * HSTRIDE + (k - 256));
      float s = so[m];
#pragma unroll
      for (int i = 0; i < 4; ++i) o[i] = (bf16_t)((float)hv[i] * s);
    } else {
      v4bf hv = *(const v4bf*)(h_self + (size_t)m * HSTRIDE + (k - 512));
      float s = ss[m];
#pragma unroll
      for (int i = 0; i < 4; ++i) o[i] = (bf16_t)((float)hv[i] * s);
    }
    *(v4bf*)(xin + (size_t)m * XSTRIDE + k) = o;
  }
}

// Per-layer GEMM context: all loop-invariant pointers/bias hoisted out of
// the 256-step scan.
struct GemmCtx {
  const bf16_t* pA;    // A row base in xin (lane row + khalf baked in)
  const bf16_t* pR;    // r-gate weight tile base (+512/kb)
  const bf16_t* pZ;    // z-gate
  const bf16_t* pGI;   // n-gate input path (K 0..511)
  const bf16_t* pGH;   // n-gate hidden path (K 512..767)
  float br, bz, bgi, bgh;
};

// GEMM (72 WMMAs/wave) + GRU pointwise math + state/output stores.
__device__ __forceinline__ void gemm_gates(
    const GemmCtx& cx,
    const bf16_t* __restrict__ xin,
    bf16_t*       __restrict__ hstore,
    const float*  __restrict__ cmask,
    float*        __restrict__ outbase,
    int col, int khalf) {
  v8f accR, accZ, accGI, accGH;
#pragma unroll
  for (int i = 0; i < 8; ++i) {
    accR[i] = cx.br; accZ[i] = cx.bz; accGI[i] = cx.bgi; accGH[i] = cx.bgh;
  }

#pragma unroll 4
  for (int kb = 0; kb < 16; ++kb) {           // K region [0,512): r, z, gi_n
    v16bf a  = ld_frag(cx.pA  + kb * 32);
    v16bf br = ld_frag(cx.pR  + kb * 512);
    v16bf bz = ld_frag(cx.pZ  + kb * 512);
    v16bf bg = ld_frag(cx.pGI + kb * 512);
    accR  = wmma_bf16(a, br, accR);
    accZ  = wmma_bf16(a, bz, accZ);
    accGI = wmma_bf16(a, bg, accGI);
  }
#pragma unroll 4
  for (int kb = 0; kb < 8; ++kb) {            // K region [512,768): r, z, gh_n
    v16bf a  = ld_frag(cx.pA + (16 + kb) * 32);
    v16bf br = ld_frag(cx.pR + (16 + kb) * 512);
    v16bf bz = ld_frag(cx.pZ + (16 + kb) * 512);
    v16bf bh = ld_frag(cx.pGH + kb * 512);
    accR  = wmma_bf16(a, br, accR);
    accZ  = wmma_bf16(a, bz, accZ);
    accGH = wmma_bf16(a, bh, accGH);
  }

  // C layout: element i <-> (row = i + 8*khalf, column = col)
#pragma unroll
  for (int i = 0; i < 8; ++i) {
    int   row = i + 8 * khalf;
    float r   = sigmoidf_(accR[i]);
    float z   = sigmoidf_(accZ[i]);
    float n   = tanhf(accGI[i] + r * accGH[i]);
    float h   = (float)xin[(size_t)row * XSTRIDE + 512 + col]; // masked h_self
    float c   = cmask[row];
    float g   = (1.0f - z) * n + z * h;
    float hn  = c * h + (1.0f - c) * g;
    hstore[(size_t)row * HSTRIDE + col] = (bf16_t)hn;
    outbase[(size_t)row * TDIM * HDIM + col] = hn;
  }
}

__global__ __launch_bounds__(512) void hmgru_main(
    const float* __restrict__ x0,  const float* __restrict__ x1,
    const float* __restrict__ hx0, const float* __restrict__ hx1,
    const int*   __restrict__ dx,  const int*   __restrict__ dx0,
    const bf16_t* __restrict__ wp, const float* __restrict__ bp,
    float* __restrict__ out)
{
  __shared__ __align__(16) bf16_t xin[16 * XSTRIDE];
  __shared__ __align__(16) bf16_t h0s[16 * HSTRIDE];
  __shared__ __align__(16) bf16_t h1s[16 * HSTRIDE];
  __shared__ __align__(16) float  xraw[16 * 256];   // async-DMA'd x tile
  __shared__ float rc[8][16];  // b0, dd0, c0, b1, dd1, c1, 1-dd0, 1-dd1

  const int tid   = threadIdx.x;
  const int btile = blockIdx.x;
  const int b0row = btile * 16;

  const int lane  = tid & 31;
  const int w     = tid >> 5;
  const int coln  = lane & 15;
  const int khalf = (lane >> 4) & 1;
  const int col   = w * 16 + coln;
  const int boff  = coln * 32 + khalf * 8;

  // Warm the bf16 weight pack into L2 (stays resident: 2.25MB << 192MB).
  {
    const char* wb = (const char*)wp;
    for (int c = tid; c < (2 * WL * 2) / 128; c += 512)
      __builtin_prefetch(wb + (size_t)c * 128, 0, 1);
  }

  // Hoist all scan-invariant GEMM state.
  GemmCtx cx0, cx1;
  {
    const bf16_t* arow = xin + (size_t)(lane & 15) * XSTRIDE + khalf * 8;
    const bf16_t* w0 = wp;
    const bf16_t* w1 = wp + WL;
    cx0.pA  = arow;
    cx0.pR  = w0 + (size_t)(w * 24) * 512 + boff;
    cx0.pZ  = w0 + (size_t)((16 + w) * 24) * 512 + boff;
    cx0.pGI = w0 + (size_t)W_RZ + (size_t)(w * 16) * 512 + boff;
    cx0.pGH = w0 + (size_t)W_RZ + W_GIN + (size_t)(w * 8) * 512 + boff;
    cx0.br  = bp[col];       cx0.bz  = bp[256 + col];
    cx0.bgi = bp[512 + col]; cx0.bgh = bp[768 + col];
    cx1.pA  = arow;
    cx1.pR  = w1 + (size_t)(w * 24) * 512 + boff;
    cx1.pZ  = w1 + (size_t)((16 + w) * 24) * 512 + boff;
    cx1.pGI = w1 + (size_t)W_RZ + (size_t)(w * 16) * 512 + boff;
    cx1.pGH = w1 + (size_t)W_RZ + W_GIN + (size_t)(w * 8) * 512 + boff;
    cx1.br  = bp[1024 + col];       cx1.bz  = bp[1024 + 256 + col];
    cx1.bgi = bp[1024 + 512 + col]; cx1.bgh = bp[1024 + 768 + col];
  }
  float* outL0 = out + ((size_t)b0row * TDIM) * HDIM;
  float* outL1 = out + (((size_t)BDIM + b0row) * TDIM) * HDIM;

  // init hidden state
  for (int e = tid; e < 16 * HDIM; e += 512) {
    int m = e >> 8, k = e & 255;
    h0s[m * HSTRIDE + k] = (bf16_t)hx0[(size_t)(b0row + m) * HDIM + k];
    h1s[m * HSTRIDE + k] = (bf16_t)hx1[(size_t)(b0row + m) * HDIM + k];
  }
  issue_x_tile(x0, b0row, 0, xraw, tid);   // prefetch x0(t=0) via DMA
  __syncthreads();

  for (int t = 0; t < TDIM; ++t) {
    wait_async();                          // xraw <- x0(t) complete (this wave)
    if (tid < 16) {
      int   b   = b0row + tid;
      float bb0 = (float)dx0[(size_t)b * TDIM + t];
      float dd0 = (t > 0) ? (float)dx[((size_t)b * 2 + 0) * TDIM + (t - 1)] : 0.0f;
      float bb1 = (float)dx[((size_t)b * 2 + 0) * TDIM + t];
      float dd1 = (t > 0) ? (float)dx[((size_t)b * 2 + 1) * TDIM + (t - 1)] : 0.0f;
      rc[0][tid] = bb0;  rc[1][tid] = dd0;
      rc[2][tid] = (bb0 + dd0 == 0.0f) ? 1.0f : 0.0f;
      rc[3][tid] = bb1;  rc[4][tid] = dd1;
      rc[5][tid] = (bb1 + dd1 == 0.0f) ? 1.0f : 0.0f;
      rc[6][tid] = 1.0f - dd0;
      rc[7][tid] = 1.0f - dd1;
    }
    __syncthreads();                       // DMA visible block-wide + rc ready

    // layer 0: xin = [x0*b0 | h1_prev*dd0 | h0_prev*(1-dd0)]
    stage_xin(xin, xraw, h1s, h0s, rc[0], rc[1], rc[6], tid);
    __syncthreads();

    issue_x_tile(x1, b0row, t, xraw, tid); // DMA x1(t) under layer-0 GEMM
    gemm_gates(cx0, xin, h0s, rc[2], outL0 + (size_t)t * HDIM, col, khalf);
    __syncthreads();                       // h0s ready, xin free

    wait_async();                          // xraw <- x1(t)
    __syncthreads();

    // layer 1: xin = [x1*b1 | h0_new*b1 | h1_prev*(1-dd1)]
    stage_xin(xin, xraw, h0s, h1s, rc[3], rc[3], rc[7], tid);
    __syncthreads();

    if (t + 1 < TDIM)
      issue_x_tile(x0, b0row, t + 1, xraw, tid); // DMA x0(t+1) under layer-1 GEMM
    gemm_gates(cx1, xin, h1s, rc[5], outL1 + (size_t)t * HDIM, col, khalf);
    __syncthreads();                       // h1s ready, xin free
  }
}

// ---------------------------------------------------------------------------
// Weight/bias prepack: f32 -> bf16, (ntile, kblock) 16x32 tiles, each
// column's 32 K-values contiguous (B-fragment order). Biases prefused.
// ---------------------------------------------------------------------------
__global__ void hmgru_pack(
    const float* __restrict__ Wih0, const float* __restrict__ Whh0,
    const float* __restrict__ bih0, const float* __restrict__ bhh0,
    const float* __restrict__ Wih1, const float* __restrict__ Whh1,
    const float* __restrict__ bih1, const float* __restrict__ bhh1,
    bf16_t* __restrict__ wp, float* __restrict__ bp)
{
  size_t idx = (size_t)blockIdx.x * blockDim.x + threadIdx.x;
  const size_t total = (size_t)2 * WL;
  if (idx < total) {
    int layer = (int)(idx / WL);
    int r     = (int)(idx % WL);
    const float* Wih = layer ? Wih1 : Wih0;
    const float* Whh = layer ? Whh1 : Whh0;
    float val;
    if (r < W_RZ) {                       // fused r/z: K = [W_ih(512) | W_hh(256)]
      int tile = r >> 9, within = r & 511;
      int n = within >> 5, kk = within & 31;
      int nt = tile / 24, kb = tile % 24;
      int row = nt * 16 + n, k = kb * 32 + kk;
      val = (k < 512) ? Wih[(size_t)row * 512 + k]
                      : Whh[(size_t)row * 256 + (k - 512)];
    } else if (r < W_RZ + W_GIN) {        // n-gate input path, W_ih rows 512..767
      int r2 = r - W_RZ;
      int tile = r2 >> 9, within = r2 & 511;
      int n = within >> 5, kk = within & 31;
      int nt = tile / 16, kb = tile % 16;
      int row = 512 + nt * 16 + n, k = kb * 32 + kk;
      val = Wih[(size_t)row * 512 + k];
    } else {                              // n-gate hidden path, W_hh rows 512..767
      int r3 = r - W_RZ - W_GIN;
      int tile = r3 >> 9, within = r3 & 511;
      int n = within >> 5, kk = within & 31;
      int nt = tile / 8, kb = tile % 8;
      int row = 512 + nt * 16 + n, k = kb * 32 + kk;
      val = Whh[(size_t)row * 256 + k];
    }
    wp[idx] = (bf16_t)val;
  }
  if (idx < 2048) {                       // biases: [brz(512)|bgin(256)|bghn(256)] x2
    int layer = (int)(idx >> 10);
    int j     = (int)(idx & 1023);
    const float* bih = layer ? bih1 : bih0;
    const float* bhh = layer ? bhh1 : bhh0;
    float v;
    if (j < 512)      v = bih[j] + bhh[j];
    else if (j < 768) v = bih[512 + (j - 512)];
    else              v = bhh[512 + (j - 768)];
    bp[(size_t)layer * 1024 + j] = v;
  }
}

extern "C" void kernel_launch(void* const* d_in, const int* in_sizes, int n_in,
                              void* d_out, int out_size, void* d_ws, size_t ws_size,
                              hipStream_t stream) {
  const float* x0   = (const float*)d_in[0];
  const float* x1   = (const float*)d_in[1];
  const float* hx0  = (const float*)d_in[2];
  const float* hx1  = (const float*)d_in[3];
  const float* Wih0 = (const float*)d_in[4];
  const float* Whh0 = (const float*)d_in[5];
  const float* bih0 = (const float*)d_in[6];
  const float* bhh0 = (const float*)d_in[7];
  const float* Wih1 = (const float*)d_in[8];
  const float* Whh1 = (const float*)d_in[9];
  const float* bih1 = (const float*)d_in[10];
  const float* bhh1 = (const float*)d_in[11];
  const int*   dx   = (const int*)d_in[12];
  const int*   dx0  = (const int*)d_in[13];
  float* out = (float*)d_out;

  // workspace: [2*WL bf16 packed weights][2*1024 f32 fused biases]
  bf16_t* wpk = (bf16_t*)d_ws;
  float*  bpk = (float*)((char*)d_ws + (size_t)2 * WL * sizeof(bf16_t));

  const int pthreads = 256;
  const int pblocks  = (2 * WL + pthreads - 1) / pthreads;
  hipLaunchKernelGGL(hmgru_pack, dim3(pblocks), dim3(pthreads), 0, stream,
                     Wih0, Whh0, bih0, bhh0, Wih1, Whh1, bih1, bhh1, wpk, bpk);
  hipLaunchKernelGGL(hmgru_main, dim3(BDIM / 16), dim3(512), 0, stream,
                     x0, x1, hx0, hx1, dx, dx0, wpk, bpk, out);
}
